// InGramRelationLayer_v2_64046552318122
// MI455X (gfx1250) — matrix operand order repaired and checked by reading, not compiled, not run on hardware
//
#include <hip/hip_runtime.h>
#include <hip/hip_bf16.h>

#define DIM 512
#define NHEAD 8

typedef __bf16 bf16_t;
typedef bf16_t v16bf __attribute__((ext_vector_type(16)));
typedef bf16_t v8bf  __attribute__((ext_vector_type(8)));
typedef float  v8f   __attribute__((ext_vector_type(8)));

union FragA { v16bf v; v8bf h[2]; };

// ---------- monotonic float <-> u32 key for atomic segment-max ----------
__device__ __forceinline__ unsigned fkey(float f) {
    unsigned u = __float_as_uint(f);
    return (u & 0x80000000u) ? ~u : (u | 0x80000000u);
}
__device__ __forceinline__ float fkeyinv(unsigned k) {
    unsigned u = (k & 0x80000000u) ? (k & 0x7FFFFFFFu) : ~k;
    return __uint_as_float(u);
}

// ---------- fp32 -> (bf16 hi, bf16 lo) with zero padding past `rows` ----------
__global__ void k_cvt_pad(const float* __restrict__ src, bf16_t* __restrict__ hi,
                          bf16_t* __restrict__ lo, int rows, int rowsPad) {
    int idx = blockIdx.x * blockDim.x + threadIdx.x;
    int total = rowsPad * DIM;
    if (idx >= total) return;
    int r = idx / DIM;
    float x = (r < rows) ? src[idx] : 0.0f;
    bf16_t h = (bf16_t)x;
    bf16_t l = (bf16_t)(x - (float)h);
    hi[idx] = h; lo[idx] = l;
}

// W_attn (512 x 1536) -> three 512x512 bf16x2 blocks (k slices 0/512/1024)
__global__ void k_cvt_wattn(const float* __restrict__ W,
                            bf16_t* h1, bf16_t* l1, bf16_t* h2, bf16_t* l2,
                            bf16_t* h3, bf16_t* l3) {
    int idx = blockIdx.x * blockDim.x + threadIdx.x;
    if (idx >= DIM * DIM) return;
    int j = idx / DIM, k = idx % DIM;
    const float* row = W + (size_t)j * (3 * DIM) + k;
    float x0 = row[0], x1 = row[DIM], x2 = row[2 * DIM];
    bf16_t a = (bf16_t)x0; h1[idx] = a; l1[idx] = (bf16_t)(x0 - (float)a);
    bf16_t b = (bf16_t)x1; h2[idx] = b; l2[idx] = (bf16_t)(x1 - (float)b);
    bf16_t c = (bf16_t)x2; h3[idx] = c; l3[idx] = (bf16_t)(x2 - (float)c);
}

__global__ void k_init(unsigned* __restrict__ segmax, float* __restrict__ segsum,
                       unsigned* __restrict__ counts, int nSeg, int nRel) {
    int idx = blockIdx.x * blockDim.x + threadIdx.x;
    if (idx < nSeg) { segmax[idx] = 0u; segsum[idx] = 0.0f; }
    if (idx < nRel) counts[idx] = 0u;
}

// ---------- C = X @ W^T (+bias), compensated bf16x2, one 16x16 tile / wave ----------
__global__ void k_gemm_bf16x2(const bf16_t* __restrict__ Xhi, const bf16_t* __restrict__ Xlo,
                              const bf16_t* __restrict__ Whi, const bf16_t* __restrict__ Wlo,
                              const float* __restrict__ bias, float* __restrict__ C,
                              int tilesM) {
    int waveId = (blockIdx.x * blockDim.x + threadIdx.x) >> 5;
    int lane   = threadIdx.x & 31;
    int totalTiles = tilesM * (DIM / 16);
    if (waveId >= totalTiles) return;
    int tileM = waveId / (DIM / 16);
    int tileN = waveId % (DIM / 16);
    int laneLo = lane & 15;
    int hiHalf = (lane >> 4) & 1;

    // A: lanes 0-15 hold K 0..15; lanes 16-31 hold K 16..31
    const size_t aBase = (size_t)(tileM * 16 + laneLo) * DIM + (hiHalf ? 16 : 0);
    // B: lane = column; lanes 0-15 hold K 0..15; lanes 16-31 hold K 16..31
    const int    col   = tileN * 16 + laneLo;
    const size_t bBase = (size_t)col * DIM + (hiHalf ? 16 : 0);

    v8f acc = {};
#pragma unroll 4
    for (int kb = 0; kb < DIM; kb += 32) {
        FragA ahi, alo;
        ahi.h[0] = *(const v8bf*)(Xhi + aBase + kb);
        ahi.h[1] = *(const v8bf*)(Xhi + aBase + kb + 8);
        alo.h[0] = *(const v8bf*)(Xlo + aBase + kb);
        alo.h[1] = *(const v8bf*)(Xlo + aBase + kb + 8);
        v16bf bhi = *(const v16bf*)(Whi + bBase + kb);
        v16bf blo = *(const v16bf*)(Wlo + bBase + kb);
        acc = __builtin_amdgcn_wmma_f32_16x16x32_bf16(false, ahi.v, false, bhi,
                                                      (short)0, acc, false, false);
        acc = __builtin_amdgcn_wmma_f32_16x16x32_bf16(false, ahi.v, false, blo,
                                                      (short)0, acc, false, false);
        acc = __builtin_amdgcn_wmma_f32_16x16x32_bf16(false, alo.v, false, bhi,
                                                      (short)0, acc, false, false);
    }
    float bv = bias ? bias[col] : 0.0f;
    int rowBase = tileM * 16 + (hiHalf ? 8 : 0);
#pragma unroll
    for (int i = 0; i < 8; i++)
        C[(size_t)(rowBase + i) * DIM + col] = acc[i] + bv;
}

// ---------- one wave per edge: attn_raw + atomic segment-max ----------
__global__ void k_edge_attn(const long long* __restrict__ trip,
                            const float* __restrict__ A1, const float* __restrict__ A2,
                            const float* __restrict__ A3, const float* __restrict__ vec,
                            float* __restrict__ ev_ws, unsigned* __restrict__ segmax,
                            int nEdges) {
    int wave = (int)((blockIdx.x * (unsigned)blockDim.x + threadIdx.x) >> 5);
    int lane = threadIdx.x & 31;
    if (wave >= nEdges) return;
    int hd = (int)trip[(size_t)wave * 3 + 0];
    int tl = (int)trip[(size_t)wave * 3 + 1];
    int bn = (int)trip[(size_t)wave * 3 + 2];
    const float* a1 = A1 + (size_t)hd * DIM;
    const float* a2 = A2 + (size_t)bn * DIM;
    const float* a3 = A3 + (size_t)tl * DIM;

    float myv = 0.0f;
#pragma unroll
    for (int hh = 0; hh < NHEAD; hh++) {
        float s = 0.0f;
#pragma unroll
        for (int j = 0; j < 2; j++) {
            int d = lane + 32 * (2 * hh + j);
            float x = a1[d] + a2[d] + a3[d];
            x = x > 0.0f ? x : 0.2f * x;                // leaky_relu(0.2)
            s += x * vec[d];
        }
#pragma unroll
        for (int off = 16; off; off >>= 1) s += __shfl_xor(s, off, 32);
        if (lane == hh) myv = s;
    }
    if (lane < NHEAD) {
        ev_ws[(size_t)wave * NHEAD + lane] = myv;
        atomicMax(&segmax[(size_t)hd * NHEAD + lane], fkey(myv));
    }
}

// ---------- ev = exp(attn - segmax), atomic segment-sum ----------
__global__ void k_edge_softmax(const long long* __restrict__ trip,
                               float* __restrict__ ev_ws,
                               const unsigned* __restrict__ segmax,
                               float* __restrict__ segsum, int nEdges) {
    int idx = blockIdx.x * blockDim.x + threadIdx.x;
    if (idx >= nEdges * NHEAD) return;
    int e = idx >> 3, hh = idx & 7;
    int hd = (int)trip[(size_t)e * 3 + 0];
    float mx = fkeyinv(segmax[(size_t)hd * NHEAD + hh]);
    float ev = __expf(ev_ws[idx] - mx);
    ev_ws[idx] = ev;
    atomicAdd(&segsum[(size_t)hd * NHEAD + hh], ev);
}

// ---------- CSR build: histogram, block scan, scatter ----------
__global__ void k_hist(const long long* __restrict__ trip, unsigned* __restrict__ counts,
                       int nEdges) {
    int e = blockIdx.x * blockDim.x + threadIdx.x;
    if (e >= nEdges) return;
    atomicAdd(&counts[(int)trip[(size_t)e * 3 + 0]], 1u);
}

__global__ void k_scan(const unsigned* __restrict__ counts, unsigned* __restrict__ offsets,
                       unsigned* __restrict__ cursor, int n) {
    __shared__ unsigned part[256];
    int t = threadIdx.x;
    int chunk = (n + 255) / 256;
    int s = t * chunk;
    int e = s + chunk < n ? s + chunk : n;
    unsigned sum = 0;
    for (int i = s; i < e; i++) sum += counts[i];
    part[t] = sum;
    __syncthreads();
    for (int off = 1; off < 256; off <<= 1) {
        unsigned v = (t >= off) ? part[t - off] : 0u;
        __syncthreads();
        part[t] += v;
        __syncthreads();
    }
    unsigned base = (t == 0) ? 0u : part[t - 1];
    for (int i = s; i < e; i++) {
        offsets[i] = base; cursor[i] = base;
        base += counts[i];
    }
    if (t == 255) offsets[n] = part[255];
}

__global__ void k_scatter(const long long* __restrict__ trip, unsigned* __restrict__ cursor,
                          unsigned* __restrict__ edge_ids, int nEdges) {
    int e = blockIdx.x * blockDim.x + threadIdx.x;
    if (e >= nEdges) return;
    unsigned pos = atomicAdd(&cursor[(int)trip[(size_t)e * 3 + 0]], 1u);
    edge_ids[pos] = (unsigned)e;
}

// ---------- one block per relation: out[r] = sum_e beta * M[tail] ----------
__global__ void k_out(const long long* __restrict__ trip, const unsigned* __restrict__ offsets,
                      const unsigned* __restrict__ edge_ids, const float* __restrict__ ev_ws,
                      const float* __restrict__ segsum, const float* __restrict__ Mrow,
                      float* __restrict__ out) {
    int r = blockIdx.x;
    int t = threadIdx.x;
    __shared__ float inv[NHEAD];
    if (t < NHEAD) inv[t] = 1.0f / (segsum[(size_t)r * NHEAD + t] + 1e-16f);
    __syncthreads();
    unsigned s = offsets[r], e = offsets[r + 1];
    int d0 = t, d1 = t + 256;
    int h0 = t >> 6, h1 = h0 + 4;
    float acc0 = 0.0f, acc1 = 0.0f;
    for (unsigned i = s; i < e; i++) {
        int ed = (int)edge_ids[i];
        int tl = (int)trip[(size_t)ed * 3 + 1];
        const float* m = Mrow + (size_t)tl * DIM;
        float b0 = ev_ws[(size_t)ed * NHEAD + h0] * inv[h0];
        float b1 = ev_ws[(size_t)ed * NHEAD + h1] * inv[h1];
        acc0 += b0 * m[d0];
        acc1 += b1 * m[d1];
    }
    out[(size_t)r * DIM + d0] = acc0;
    out[(size_t)r * DIM + d1] = acc1;
}

extern "C" void kernel_launch(void* const* d_in, const int* in_sizes, int n_in,
                              void* d_out, int out_size, void* d_ws, size_t ws_size,
                              hipStream_t stream) {
    const float*     emb_rel = (const float*)d_in[0];
    const float*     emb_bin = (const float*)d_in[1];
    const long long* trip    = (const long long*)d_in[2];
    const float*     W_attn  = (const float*)d_in[3];
    const float*     b_attn  = (const float*)d_in[4];
    const float*     attnvec = (const float*)d_in[5];
    const float*     W_aggr  = (const float*)d_in[6];
    const float*     b_aggr  = (const float*)d_in[7];
    float* out = (float*)d_out;

    const int nRel   = in_sizes[0] / DIM;            // 5000
    const int nBin   = in_sizes[1] / DIM;            // 32
    const int nEdges = in_sizes[2] / 3;              // 200000
    const int nRelPad = (nRel + 15) & ~15;           // 5008
    const int nBinPad = (nBin + 15) & ~15;           // 32

    // workspace carve-out (256B aligned slices)
    char* p = (char*)d_ws;
    auto take = [&](size_t bytes) -> void* {
        void* r = (void*)p;
        p += (bytes + 255) & ~(size_t)255;
        return r;
    };
    bf16_t* relHi = (bf16_t*)take((size_t)nRelPad * DIM * 2);
    bf16_t* relLo = (bf16_t*)take((size_t)nRelPad * DIM * 2);
    bf16_t* binHi = (bf16_t*)take((size_t)nBinPad * DIM * 2);
    bf16_t* binLo = (bf16_t*)take((size_t)nBinPad * DIM * 2);
    bf16_t* w1Hi  = (bf16_t*)take((size_t)DIM * DIM * 2);
    bf16_t* w1Lo  = (bf16_t*)take((size_t)DIM * DIM * 2);
    bf16_t* w2Hi  = (bf16_t*)take((size_t)DIM * DIM * 2);
    bf16_t* w2Lo  = (bf16_t*)take((size_t)DIM * DIM * 2);
    bf16_t* w3Hi  = (bf16_t*)take((size_t)DIM * DIM * 2);
    bf16_t* w3Lo  = (bf16_t*)take((size_t)DIM * DIM * 2);
    bf16_t* waHi  = (bf16_t*)take((size_t)DIM * DIM * 2);
    bf16_t* waLo  = (bf16_t*)take((size_t)DIM * DIM * 2);
    float* A1 = (float*)take((size_t)nRelPad * DIM * 4);
    float* A2 = (float*)take((size_t)nBinPad * DIM * 4);
    float* A3 = (float*)take((size_t)nRelPad * DIM * 4);
    float* Mr = (float*)take((size_t)nRelPad * DIM * 4);
    float*    ev_ws  = (float*)take((size_t)nEdges * NHEAD * 4);
    unsigned* segmax = (unsigned*)take((size_t)nRel * NHEAD * 4);
    float*    segsum = (float*)take((size_t)nRel * NHEAD * 4);
    unsigned* counts  = (unsigned*)take((size_t)nRel * 4);
    unsigned* offsets = (unsigned*)take((size_t)(nRel + 1) * 4);
    unsigned* cursor  = (unsigned*)take((size_t)nRel * 4);
    unsigned* edge_ids = (unsigned*)take((size_t)nEdges * 4);

    const int B = 256;
    auto grid = [](long long n, int b) { return (unsigned)((n + b - 1) / b); };

    // 1) precision-split conversions
    k_cvt_pad<<<grid((long long)nRelPad * DIM, B), B, 0, stream>>>(emb_rel, relHi, relLo, nRel, nRelPad);
    k_cvt_pad<<<grid((long long)nBinPad * DIM, B), B, 0, stream>>>(emb_bin, binHi, binLo, nBin, nBinPad);
    k_cvt_wattn<<<grid((long long)DIM * DIM, B), B, 0, stream>>>(W_attn, w1Hi, w1Lo, w2Hi, w2Lo, w3Hi, w3Lo);
    k_cvt_pad<<<grid((long long)DIM * DIM, B), B, 0, stream>>>(W_aggr, waHi, waLo, DIM, DIM);
    k_init<<<grid((long long)nRel * NHEAD, B), B, 0, stream>>>(segmax, segsum, counts, nRel * NHEAD, nRel);

    // 2) WMMA GEMMs (one 16x16 tile per wave, 8 waves per block)
    int tilesRel = nRelPad / 16, tilesBin = nBinPad / 16;
    unsigned gRel = grid((long long)tilesRel * (DIM / 16) * 32, B);
    unsigned gBin = grid((long long)tilesBin * (DIM / 16) * 32, B);
    k_gemm_bf16x2<<<gRel, B, 0, stream>>>(relHi, relLo, w1Hi, w1Lo, b_attn, A1, tilesRel);
    k_gemm_bf16x2<<<gBin, B, 0, stream>>>(binHi, binLo, w2Hi, w2Lo, nullptr, A2, tilesBin);
    k_gemm_bf16x2<<<gRel, B, 0, stream>>>(relHi, relLo, w3Hi, w3Lo, nullptr, A3, tilesRel);
    k_gemm_bf16x2<<<gRel, B, 0, stream>>>(relHi, relLo, waHi, waLo, b_aggr, Mr, tilesRel);

    // 3) per-edge attention + segment softmax (L2-resident gathers)
    k_edge_attn<<<grid((long long)nEdges * 32, B), B, 0, stream>>>(trip, A1, A2, A3, attnvec,
                                                                   ev_ws, segmax, nEdges);
    k_edge_softmax<<<grid((long long)nEdges * NHEAD, B), B, 0, stream>>>(trip, ev_ws, segmax,
                                                                         segsum, nEdges);

    // 4) CSR by head, then one-block-per-relation aggregation (no output atomics)
    k_hist<<<grid(nEdges, B), B, 0, stream>>>(trip, counts, nEdges);
    k_scan<<<1, B, 0, stream>>>(counts, offsets, cursor, nRel);
    k_scatter<<<grid(nEdges, B), B, 0, stream>>>(trip, cursor, edge_ids, nEdges);
    k_out<<<(unsigned)nRel, B, 0, stream>>>(trip, offsets, edge_ids, ev_ws, segsum, Mr, out);
}
